// MomentLayer_10179072491907
// MI455X (gfx1250) — matrix operand order repaired and checked by reading, not compiled, not run on hardware
//
#include <hip/hip_runtime.h>
#include <math.h>

typedef __bf16 bf16;
typedef __attribute__((ext_vector_type(16))) __bf16 v16bf;
typedef __attribute__((ext_vector_type(8)))  float  v8f;
typedef unsigned int v4u __attribute__((ext_vector_type(4)));
typedef int          v8i __attribute__((ext_vector_type(8)));
typedef int          v4i __attribute__((ext_vector_type(4)));

#define B_DIM 512
#define N_DIM 256

// LDS leading dims (bf16 elements), padded so 16-lane b128 fragment loads are bank-conflict free
constexpr int WBLD = 264;   // Wb rows: 256 cols + 8 pad (row stride 528B -> bank advance 4)
constexpr int RTLD = 264;   // transposed R chunk rows
constexpr int T1LD = 40;    // T1 rows: 32 cols + 8 pad (row stride 80B -> bank advance 20)
constexpr int STAGE_OFF  = (256 * WBLD + 32 * RTLD + 256 * T1LD) * 2 + 2 * 256 * 4; // 174592
constexpr int STAGE_BYTES = 256 * 32 * 4;            // one fp32 R chunk (32KB)
constexpr int SMEM_BYTES  = STAGE_OFF + 2 * STAGE_BYTES; // double-buffered TDM staging

union Frag32B { uint4 q[2]; v16bf v; };

// A-fragment 16x32 (MxK) from row-major LDS [row][k].
// Lane L<16: row=rowBase+L, K = {0..7, 16..23}; lane L>=16: row=rowBase+(L-16), K = {8..15, 24..31}
__device__ __forceinline__ v16bf load_frag_a(const bf16* lds, int rowBase, int kBase, int ld, int lane) {
  const bf16* p = lds + (size_t)(rowBase + (lane & 15)) * ld + kBase + ((lane < 16) ? 0 : 8);
  Frag32B f;
  f.q[0] = *(const uint4*)(p);
  f.q[1] = *(const uint4*)(p + 16);
  return f.v;
}

// B-fragment 32x16 (KxN) from LDS stored as [n][k] row-major (i.e. column-major B).
// Lane L<16: col=nBase+L, K = {0..7, 16..23}; lane L>=16: col=nBase+(L-16), K = {8..15, 24..31}
__device__ __forceinline__ v16bf load_frag_b(const bf16* lds, int nBase, int kBase, int ld, int lane) {
  const bf16* p = lds + (size_t)(nBase + (lane & 15)) * ld + kBase + ((lane < 16) ? 0 : 8);
  Frag32B f;
  f.q[0] = *(const uint4*)(p);
  f.q[1] = *(const uint4*)(p + 16);
  return f.v;
}

// ---- Tensor Data Mover: async DMA of a 2D fp32 tile (32 wide x 256 rows, row stride 256) into LDS ----
__device__ __forceinline__ void tdm_load_rho_chunk(const float* gsrc, unsigned lds_off) {
  unsigned long long ga = (unsigned long long)(uintptr_t)gsrc;
  v4u g0;
  g0[0] = 1u;                                            // count=1 (valid user descriptor)
  g0[1] = lds_off;                                       // lds_addr (bytes, wave-relative)
  g0[2] = (unsigned)(ga & 0xffffffffu);                  // global_addr[31:0]
  g0[3] = (unsigned)((ga >> 32) & 0x01ffffffu) | 0x80000000u; // global_addr[56:32] | type=2
  v8i g1;
  g1[0] = (int)(2u << 16);      // workgroup_mask=0, data_size=2 (4 bytes)
  g1[1] = (int)(256u << 16);    // atomic_barrier_addr=0, tensor_dim0[15:0]=256
  g1[2] = (int)(256u << 16);    // tensor_dim0[31:16]=0, tensor_dim1[15:0]=256
  g1[3] = (int)(32u << 16);     // tensor_dim1[31:16]=0, tile_dim0=32
  g1[4] = (int)(256u);          // tile_dim1=256, tile_dim2=0
  g1[5] = (int)(256u);          // tensor_dim0_stride[31:0]=256
  g1[6] = 0;                    // tensor_dim0_stride[47:32]=0, tensor_dim1_stride lo=0
  g1[7] = 0;
  v4i gz = {0, 0, 0, 0};
#if defined(__clang_major__) && (__clang_major__ >= 23)
  v8i gz8 = {0, 0, 0, 0, 0, 0, 0, 0};
  __builtin_amdgcn_tensor_load_to_lds(g0, g1, gz, gz, gz8, 0);
#else
  __builtin_amdgcn_tensor_load_to_lds(g0, g1, gz, gz, 0);
#endif
}

// ---------------- Kernel 1: u2 = u @ W^T + b ----------------
__global__ __launch_bounds__(256) void linear_mean_kernel(
    const float* __restrict__ u, const float* __restrict__ W,
    const float* __restrict__ bias, float* __restrict__ u2) {
  __shared__ float uRow[N_DIM];
  const int b = blockIdx.x, o = threadIdx.x;
  uRow[o] = u[b * N_DIM + o];
  __syncthreads();
  float acc = bias[o];
  const float* wr = W + (size_t)o * N_DIM;
  for (int i = 0; i < N_DIM; ++i) acc = fmaf(uRow[i], wr[i], acc);
  u2[b * N_DIM + o] = acc;
}

// ---------------- Kernel 2: BatchNorm stats over batch dim ----------------
__global__ __launch_bounds__(256) void bn_stats_kernel(
    const float* __restrict__ u2, const float* __restrict__ gamma,
    float* __restrict__ mu_out, float* __restrict__ inv_out) {
  const int n = threadIdx.x;
  float sum = 0.f, sumsq = 0.f;
  for (int b = 0; b < B_DIM; ++b) {
    float v = u2[b * N_DIM + n];
    sum += v; sumsq += v * v;
  }
  float mu  = sum * (1.f / B_DIM);
  float var = sumsq * (1.f / B_DIM) - mu * mu;
  mu_out[n]  = mu;
  inv_out[n] = gamma[n] * rsqrtf(var + 1e-5f);
}

// ---------------- Kernel 3: per-batch cov2 = Wb R Wb^T via bf16 WMMA + TDM double buffering ----------------
// One workgroup per batch, 32 waves, ~235KB LDS (1 WG/WGP -> TDM prefetch hides HBM latency).
// Wave 0 DMAs R chunk jc+1 into staging[(jc+1)&1] while all waves compute chunk jc.
__global__ __launch_bounds__(1024) void cov_wmma_kernel(
    const float* __restrict__ W, const float* __restrict__ s,
    const float* __restrict__ rho, float* __restrict__ rho2_out,
    float* __restrict__ s2_out) {
  extern __shared__ char smem_raw[];
  bf16*  Wb    = (bf16*)smem_raw;                  // [256][WBLD]
  bf16*  Rt    = Wb + 256 * WBLD;                  // [32][RTLD]  (R chunk transposed bf16: Rt[j][i])
  bf16*  T1    = Rt + 32 * RTLD;                   // [256][T1LD]
  float* sRow  = (float*)(T1 + 256 * T1LD);
  float* s2Row = sRow + 256;
  float* stage = (float*)(smem_raw + STAGE_OFF);   // 2 x [256][32] fp32 TDM staging

  const int b    = blockIdx.x;
  const int tid  = threadIdx.x;
  const int w    = tid >> 5;
  const int lane = tid & 31;

  const size_t rhoBase = (size_t)b * (N_DIM * N_DIM);
  const unsigned ldsBase = (unsigned)(uintptr_t)(void*)smem_raw; // low 32 bits = LDS byte offset

  // kick off DMA of chunk 0 as early as possible
  if (w == 0) tdm_load_rho_chunk(rho + rhoBase, ldsBase + STAGE_OFF);

  if (tid < N_DIM) sRow[tid] = s[b * N_DIM + tid];
  __syncthreads();

  // Wb[o][i] = W[o][i] * s[b][i]  (bf16)
  for (int e = tid; e < N_DIM * N_DIM; e += 1024) {
    int i = e & 255;
    Wb[(e >> 8) * WBLD + i] = (bf16)(W[e] * sRow[i]);
  }

  const int om1 = w & 15, jn1 = w >> 4;      // stage1 tile: (om1, jn1)
  const int om2 = w >> 1, pnB = (w & 1) * 8; // stage2 tiles: (om2, pnB..pnB+7)

  v8f acc[8];
  {
    v8f z = {0.f, 0.f, 0.f, 0.f, 0.f, 0.f, 0.f, 0.f};
#pragma unroll
    for (int t = 0; t < 8; ++t) acc[t] = z;
  }

  for (int jc = 0; jc < 8; ++jc) {
    const int j0 = jc * 32;
    // prefetch next chunk, then wait for current (TDM ops complete in order per wave)
    if (w == 0) {
      if (jc < 7) {
        tdm_load_rho_chunk(rho + rhoBase + (j0 + 32),
                           ldsBase + STAGE_OFF + (unsigned)(((jc + 1) & 1) * STAGE_BYTES));
        __builtin_amdgcn_s_wait_tensorcnt(1);
      } else {
        __builtin_amdgcn_s_wait_tensorcnt(0);
      }
    }
    __syncthreads();   // chunk jc resident; also fences prior-iteration LDS readers

    // ---- convert staging (fp32 [i][32]) -> Rt (bf16 [j][i]) ----
    {
      const float* sb = stage + (jc & 1) * (STAGE_BYTES / 4);
      int i  = tid >> 2;
      int jg = (tid & 3) * 8;
      const float* gp = sb + i * 32 + jg;
      float4 v0 = *(const float4*)(gp);
      float4 v1 = *(const float4*)(gp + 4);
      bf16 tmp[8] = {(bf16)v0.x, (bf16)v0.y, (bf16)v0.z, (bf16)v0.w,
                     (bf16)v1.x, (bf16)v1.y, (bf16)v1.z, (bf16)v1.w};
#pragma unroll
      for (int k = 0; k < 8; ++k) Rt[(jg + k) * RTLD + i] = tmp[k];
    }
    __syncthreads();

    // ---- stage 1: T1 tile (om1, jn1), K = i over 8 chunks of 32 ----
    {
      v8f c1 = {0.f, 0.f, 0.f, 0.f, 0.f, 0.f, 0.f, 0.f};
#pragma unroll
      for (int ic = 0; ic < 8; ++ic) {
        v16bf a  = load_frag_a(Wb, om1 * 16, ic * 32, WBLD, lane);
        v16bf bb = load_frag_b(Rt, jn1 * 16, ic * 32, RTLD, lane);
        c1 = __builtin_amdgcn_wmma_f32_16x16x32_bf16(false, a, false, bb, (short)0, c1, false, false);
      }
      const int hi  = (lane < 16) ? 0 : 8;
      const int col = jn1 * 16 + (lane & 15);
#pragma unroll
      for (int r = 0; r < 8; ++r)
        T1[(om1 * 16 + r + hi) * T1LD + col] = (bf16)c1[r];
    }
    __syncthreads();

    // ---- stage 2: rank-32 update, K = jj (one chunk of 32) ----
    {
      v16bf a2 = load_frag_a(T1, om2 * 16, 0, T1LD, lane);
#pragma unroll
      for (int t = 0; t < 8; ++t) {
        v16bf b2 = load_frag_b(Wb, (pnB + t) * 16, j0, WBLD, lane);
        acc[t] = __builtin_amdgcn_wmma_f32_16x16x32_bf16(false, a2, false, b2, (short)0, acc[t], false, false);
      }
    }
  }
  __syncthreads();

  // ---- diagonal extraction: s2 = sqrt(max(diag,0)) ----
  if (om2 >= pnB && om2 < pnB + 8) {
    const int t  = om2 - pnB;
    const int hi = (lane < 16) ? 0 : 8;
#pragma unroll
    for (int r = 0; r < 8; ++r) {
      int m = r + hi;
      if ((lane & 15) == m) {      // this lane's column == row -> diagonal element
        float s2v = sqrtf(fmaxf(acc[t][r], 0.f));
        int o = om2 * 16 + m;
        s2Row[o] = s2v;
        s2_out[b * N_DIM + o] = s2v;
      }
    }
  }
  __syncthreads();

  // ---- normalize to rho2 and write out ----
  {
    const int hi   = (lane < 16) ? 0 : 8;
    const int colL = lane & 15;
#pragma unroll
    for (int t = 0; t < 8; ++t) {
      const int p = (pnB + t) * 16 + colL;
      const float s2p = s2Row[p];
#pragma unroll
      for (int r = 0; r < 8; ++r) {
        const int o = om2 * 16 + r + hi;
        rho2_out[rhoBase + (size_t)o * N_DIM + p] = acc[t][r] / (s2Row[o] * s2p + 1e-12f);
      }
    }
  }
}

// ---------------- Kernel 4: LIF moment activation (quadrature) ----------------
__constant__ float GLN[16] = {
  -0.98940093f, -0.94457502f, -0.86563120f, -0.75540441f,
  -0.61787624f, -0.45801678f, -0.28160355f, -0.09501251f,
   0.09501251f,  0.28160355f,  0.45801678f,  0.61787624f,
   0.75540441f,  0.86563120f,  0.94457502f,  0.98940093f };
__constant__ float GLW[16] = {
  0.02715246f, 0.07134433f, 0.10094204f, 0.12462897f,
  0.14959599f, 0.16915652f, 0.18260341f, 0.18945061f,
  0.18945061f, 0.18260341f, 0.16915652f, 0.14959599f,
  0.12462897f, 0.10094204f, 0.07134433f, 0.02715246f };

#define SQRT_PI_HALF 0.88622693f
#define SQRT_L       0.22360680f

__device__ __forceinline__ float g_fun(float x) {
  return __expf(x * x) * (SQRT_PI_HALF * (1.f + erff(x)));
}

__global__ __launch_bounds__(256) void activation_kernel(
    const float* __restrict__ u2, const float* __restrict__ mu,
    const float* __restrict__ inv, const float* __restrict__ beta,
    const float* __restrict__ s2, float* __restrict__ ua_out,
    float* __restrict__ sa_out, float* __restrict__ chi_out) {
  const int idx = blockIdx.x * 256 + threadIdx.x;
  const int n = idx & (N_DIM - 1);
  const float invn = inv[n];
  const float u  = (u2[idx] - mu[n]) * invn + beta[n];
  const float sv = s2[idx] * fabsf(invn);
  const float ss = fmaxf(sv, 1e-6f);
  const float den = 1.f / (ss * SQRT_L);
  const float ub = fminf(fmaxf((1.0f - u) * den, -4.f), 4.f);  // V_TH*L = 1.0, V_RES = 0
  const float lb = fminf(fmaxf((-u) * den, -4.f), 4.f);
  const float mid = 0.5f * (ub + lb), half = 0.5f * (ub - lb);

  float I1 = 0.f, V = 0.f;
#pragma unroll 1
  for (int q = 0; q < 16; ++q) {
    float x   = mid + half * GLN[q];
    float ex2 = __expf(x * x);
    I1 += GLW[q] * ex2 * (SQRT_PI_HALF * (1.f + erff(x)));
    float hm = 0.5f * (x + 4.f);
    float tm = 0.5f * (x - 4.f);
    float hs = 0.f;
#pragma unroll
    for (int r = 0; r < 16; ++r) {
      float t = tm + hm * GLN[r];
      float e = SQRT_PI_HALF * (1.f + erff(t));
      hs += GLW[r] * __expf(t * t) * e * e;
    }
    V += GLW[q] * ex2 * (hm * hs);
  }
  I1 *= half; V *= half;
  const float ua   = 1.f / (5.0f + 40.0f * I1);      // T_REF=5, 2/L=40
  const float varT = 3200.0f * V;                    // 8/L^2
  const float sa   = sqrtf(fmaxf(ua * ua * ua * varT, 0.f));
  const float chi  = ua * ua * 40.0f * (g_fun(ub) - g_fun(lb)) / (SQRT_L * fmaxf(sa, 1e-6f));
  ua_out[idx] = ua;
  sa_out[idx] = sa;
  chi_out[idx] = chi;
}

// ---------------- Kernel 5: rho_a = rho2 * chi_o * chi_p, diag = 1 (in place) ----------------
__global__ __launch_bounds__(256) void rho_scale_kernel(
    const float* __restrict__ chi, float* __restrict__ rho_io) {
  const int o = blockIdx.x, b = blockIdx.y, p = threadIdx.x;
  const size_t base = ((size_t)b * N_DIM + o) * N_DIM;
  float v;
  if (p == o) v = 1.0f;
  else v = rho_io[base + p] * chi[b * N_DIM + o] * chi[b * N_DIM + p];
  rho_io[base + p] = v;
}

extern "C" void kernel_launch(void* const* d_in, const int* in_sizes, int n_in,
                              void* d_out, int out_size, void* d_ws, size_t ws_size,
                              hipStream_t stream) {
  const float* u     = (const float*)d_in[0];
  const float* s     = (const float*)d_in[1];
  const float* rho   = (const float*)d_in[2];
  const float* W     = (const float*)d_in[3];
  const float* bias  = (const float*)d_in[4];
  const float* gamma = (const float*)d_in[5];
  const float* beta  = (const float*)d_in[6];

  float* out     = (float*)d_out;
  float* ua_out  = out;
  float* sa_out  = out + B_DIM * N_DIM;
  float* rho_out = out + 2 * B_DIM * N_DIM;

  float* ws   = (float*)d_ws;
  float* u2   = ws;                       // B*N
  float* s2w  = ws + 1 * B_DIM * N_DIM;   // B*N
  float* chiw = ws + 2 * B_DIM * N_DIM;   // B*N
  float* muw  = ws + 3 * B_DIM * N_DIM;   // N
  float* invw = muw + N_DIM;              // N

  hipFuncSetAttribute((const void*)cov_wmma_kernel,
                      hipFuncAttributeMaxDynamicSharedMemorySize, SMEM_BYTES);

  linear_mean_kernel<<<B_DIM, N_DIM, 0, stream>>>(u, W, bias, u2);
  bn_stats_kernel<<<1, N_DIM, 0, stream>>>(u2, gamma, muw, invw);
  cov_wmma_kernel<<<B_DIM, 1024, SMEM_BYTES, stream>>>(W, s, rho, rho_out, s2w);
  activation_kernel<<<B_DIM, N_DIM, 0, stream>>>(u2, muw, invw, beta, s2w, ua_out, sa_out, chiw);
  rho_scale_kernel<<<dim3(N_DIM, B_DIM), N_DIM, 0, stream>>>(chiw, rho_out);
}
